// Head_82051055222842
// MI455X (gfx1250) — matrix operand order repaired and checked by reading, not compiled, not run on hardware
//
#include <hip/hip_runtime.h>

typedef __attribute__((ext_vector_type(16))) _Float16 v16h;
typedef __attribute__((ext_vector_type(8)))  _Float16 v8h;
typedef __attribute__((ext_vector_type(8)))  float    v8f;

union V16U { v16h v; v8h h[2]; };

#define WMMA_F32_F16(a, b, c) \
  __builtin_amdgcn_wmma_f32_16x16x32_f16(false, (a), false, (b), (short)0, (c), false, false)

static constexpr int Bb = 8, Tt = 2048, Cc = 1024, Hh = 64;
static constexpr int WPAD = 56;   // W-stage row pitch (halves): 112B, 16B-aligned, conflict-free
static constexpr int VPAD = 24;   // v-transpose row pitch (halves): 48B, 16B-aligned
static constexpr int PPAD = 72;   // P-buffer row pitch (halves): 144B, 16B-aligned

// ---------------------------------------------------------------------------
// Kernel 1: q/k/v projections.  (B*T, C) x (C, H) -> f16.
// q,k stored [B*T, H] row-major; v stored TRANSPOSED as vT[b][h][t] so the
// attention kernel's P@V B-fragments are contiguous 32B reads (no LDS stage).
// One wave per 16-row tile; W chunk (32x64 x3) staged transposed in LDS.
// ---------------------------------------------------------------------------
__global__ __launch_bounds__(256) void proj_qkv(
    const float* __restrict__ x,
    const float* __restrict__ Wk, const float* __restrict__ Wq, const float* __restrict__ Wv,
    _Float16* __restrict__ qo, _Float16* __restrict__ ko, _Float16* __restrict__ vTo)
{
  __shared__ __align__(16) _Float16 Wt[3][64][WPAD];   // [mat][h][c-within-chunk]
  __shared__ __align__(16) _Float16 VtS[8][64][VPAD];  // per-wave v-tile transpose buffer

  const int tid    = threadIdx.x;
  const int wave   = tid >> 5;
  const int lane   = tid & 31;
  const int half16 = lane >> 4;
  const int l16    = lane & 15;
  const size_t tileRow = ((size_t)blockIdx.x * 8 + wave) * 16;   // row in [0, B*T)

  v8f acc[3][4];
  #pragma unroll
  for (int m = 0; m < 3; ++m)
    #pragma unroll
    for (int nc = 0; nc < 4; ++nc)
      #pragma unroll
      for (int r = 0; r < 8; ++r) acc[m][nc][r] = 0.0f;

  for (int kc = 0; kc < Cc; kc += 32) {
    __syncthreads();
    // stage W chunk transposed: Wt[m][h][c] = W[kc+c][h]  (coalesced reads)
    #pragma unroll
    for (int m = 0; m < 3; ++m) {
      const float* W = (m == 0) ? Wk : ((m == 1) ? Wq : Wv);
      for (int i = tid; i < 32 * 64; i += 256) {
        int kr  = i >> 6;
        int col = i & 63;
        Wt[m][col][kr] = (_Float16)W[(size_t)(kc + kr) * Hh + col];
      }
    }
    __syncthreads();

    // A fragment of x, f32 -> f16 (lanes 0-15: K 0..15; lanes 16-31: K 16..31)
    v16h a;
    {
      const float* xp = x + (tileRow + l16) * Cc + kc + half16 * 16;
      #pragma unroll
      for (int j = 0; j < 8; ++j) a[j]     = (_Float16)xp[j];
      #pragma unroll
      for (int j = 0; j < 8; ++j) a[8 + j] = (_Float16)xp[8 + j];
    }

    #pragma unroll
    for (int m = 0; m < 3; ++m) {
      #pragma unroll
      for (int nc = 0; nc < 4; ++nc) {
        V16U bu;
        const _Float16* wp = &Wt[m][nc * 16 + l16][half16 * 16];
        bu.h[0] = *(const v8h*)(wp);
        bu.h[1] = *(const v8h*)(wp + 8);
        acc[m][nc] = WMMA_F32_F16(a, bu.v, acc[m][nc]);
      }
    }
  }

  // q, k: row-major f16 stores (C-layout: row = r + half*8, col = nc*16 + l16)
  #pragma unroll
  for (int m = 0; m < 2; ++m) {
    _Float16* outp = (m == 0) ? ko : qo;
    #pragma unroll
    for (int nc = 0; nc < 4; ++nc)
      #pragma unroll
      for (int r = 0; r < 8; ++r) {
        size_t row = tileRow + r + half16 * 8;
        outp[row * Hh + nc * 16 + l16] = (_Float16)acc[m][nc][r];
      }
  }

  // v: transpose 16x64 tile through wave-private LDS, then coalesced stores
  #pragma unroll
  for (int nc = 0; nc < 4; ++nc)
    #pragma unroll
    for (int r = 0; r < 8; ++r)
      VtS[wave][nc * 16 + l16][r + half16 * 8] = (_Float16)acc[2][nc][r];

  {
    const size_t bb = tileRow / Tt;          // batch (tiles never straddle batches)
    const size_t t0 = tileRow % Tt;
    #pragma unroll
    for (int hh = lane; hh < 64; hh += 32) {
      v8h d0 = *(const v8h*)&VtS[wave][hh][0];
      v8h d1 = *(const v8h*)&VtS[wave][hh][8];
      _Float16* dst = vTo + (bb * Hh + hh) * Tt + t0;
      *(v8h*)(dst)     = d0;
      *(v8h*)(dst + 8) = d1;
    }
  }
}

// ---------------------------------------------------------------------------
// Kernel 2: causal flash attention, barrier-free.  One wave per 16-row query
// tile, streaming 64-key blocks with online softmax (base-2 domain).
// K^T and V B-fragments are contiguous 32B global reads (L2-resident);
// LDS used only for the wave-private P C-layout -> A-layout bounce.
// ---------------------------------------------------------------------------
__global__ __launch_bounds__(256) void attn_flash(
    const _Float16* __restrict__ q, const _Float16* __restrict__ k,
    const _Float16* __restrict__ vT, float* __restrict__ out)
{
  __shared__ __align__(16) _Float16 Pb[8][16][PPAD];   // per-wave 16x64 P tile

  const int tid    = threadIdx.x;
  const int wave   = tid >> 5;
  const int lane   = tid & 31;
  const int half16 = lane >> 4;
  const int l16    = lane & 15;

  const int g     = blockIdx.x * 8 + wave;     // query tile id, 0..1023
  const int b     = g >> 7;                    // 128 tiles per batch
  const int qbase = (g & 127) * 16;
  const size_t rowB = (size_t)b * Tt;
  const _Float16* vTb = vT + (size_t)b * Hh * Tt;

  // Q A-fragments (h split into two 32-chunks)
  v16h aQ[2];
  #pragma unroll
  for (int f = 0; f < 2; ++f) {
    const _Float16* qp = q + (rowB + qbase + l16) * Hh + f * 32 + half16 * 16;
    V16U u;
    u.h[0] = *(const v8h*)(qp);
    u.h[1] = *(const v8h*)(qp + 8);
    aQ[f] = u.v;
  }

  v8f accO[4];
  #pragma unroll
  for (int nc = 0; nc < 4; ++nc)
    #pragma unroll
    for (int r = 0; r < 8; ++r) accO[nc][r] = 0.0f;

  float mrow[8], lrow[8];
  #pragma unroll
  for (int r = 0; r < 8; ++r) { mrow[r] = -1e30f; lrow[r] = 0.0f; }

  const float sc2   = 0.125f * 1.44269504088896340736f;  // H^-0.5 * log2(e)
  const int   lastq = qbase + 15;
  const int   nb    = (lastq >> 6) + 1;                  // 64-key blocks

  for (int j = 0; j < nb; ++j) {
    const int kb = j * 64;
    const int ntiles = min(4, ((lastq - kb) >> 4) + 1);  // active 16-key sub-tiles

    if (j + 1 < nb) {   // hint next K block toward L2/L0 (global_prefetch_b8)
      __builtin_prefetch((const void*)(k + (rowB + kb + 64 + lane) * Hh), 0, 1);
      __builtin_prefetch((const void*)(k + (rowB + kb + 96 + lane) * Hh), 0, 1);
    }

    // S = Q @ K^T per active sub-tile (wave-uniform branch: EXEC stays full)
    float s[4][8];
    #pragma unroll
    for (int t = 0; t < 4; ++t) {
      if (t < ntiles) {
        v8f St;
        #pragma unroll
        for (int r = 0; r < 8; ++r) St[r] = 0.0f;
        #pragma unroll
        for (int c = 0; c < 2; ++c) {
          const _Float16* kp = k + (rowB + kb + t * 16 + l16) * Hh + c * 32 + half16 * 16;
          V16U bu; bu.h[0] = *(const v8h*)(kp); bu.h[1] = *(const v8h*)(kp + 8);
          St = WMMA_F32_F16(aQ[c], bu.v, St);
        }
        const int colg = kb + t * 16 + l16;
        #pragma unroll
        for (int r = 0; r < 8; ++r) {
          int rowg = qbase + r + half16 * 8;
          s[t][r] = (colg <= rowg) ? St[r] * sc2 : -1e30f;   // base-2 domain
        }
      } else {
        #pragma unroll
        for (int r = 0; r < 8; ++r) s[t][r] = -1e30f;
      }
    }

    // online softmax: rows live in 16-lane halves -> width-16 butterflies
    #pragma unroll
    for (int r = 0; r < 8; ++r) {
      float mx = fmaxf(fmaxf(s[0][r], s[1][r]), fmaxf(s[2][r], s[3][r]));
      #pragma unroll
      for (int off = 1; off < 16; off <<= 1)
        mx = fmaxf(mx, __shfl_xor(mx, off, 16));
      float mnew = fmaxf(mrow[r], mx);
      float corr = exp2f(mrow[r] - mnew);
      mrow[r] = mnew;
      float rs = 0.0f;
      #pragma unroll
      for (int t = 0; t < 4; ++t) {
        float p = exp2f(s[t][r] - mnew);    // masked/inactive -> 0
        s[t][r] = p;
        rs += p;
      }
      #pragma unroll
      for (int off = 1; off < 16; off <<= 1)
        rs += __shfl_xor(rs, off, 16);
      lrow[r] = lrow[r] * corr + rs;
      #pragma unroll
      for (int nc = 0; nc < 4; ++nc) accO[nc][r] *= corr;
    }

    // re-layout P (C-layout -> A-layout) through wave-private LDS tile
    #pragma unroll
    for (int r = 0; r < 8; ++r)
      #pragma unroll
      for (int t = 0; t < 4; ++t)
        Pb[wave][r + half16 * 8][t * 16 + l16] = (_Float16)s[t][r];

    v16h aP[2];
    #pragma unroll
    for (int f = 0; f < 2; ++f) {
      const _Float16* pp = &Pb[wave][l16][f * 32 + half16 * 16];
      V16U u; u.h[0] = *(const v8h*)(pp); u.h[1] = *(const v8h*)(pp + 8);
      aP[f] = u.v;
    }

    // O += P @ V (V B-frags contiguous from transposed vT; skip empty chunk)
    const int chunks = (ntiles > 2) ? 2 : 1;
    #pragma unroll
    for (int nc = 0; nc < 4; ++nc) {
      #pragma unroll
      for (int c = 0; c < 2; ++c) {
        if (c < chunks) {
          const _Float16* vp = vTb + (size_t)(nc * 16 + l16) * Tt + kb + c * 32 + half16 * 16;
          V16U bu; bu.h[0] = *(const v8h*)(vp); bu.h[1] = *(const v8h*)(vp + 8);
          accO[nc] = WMMA_F32_F16(aP[c], bu.v, accO[nc]);
        }
      }
    }
  }

  // normalize and store fp32 output
  #pragma unroll
  for (int nc = 0; nc < 4; ++nc)
    #pragma unroll
    for (int r = 0; r < 8; ++r) {
      size_t row = rowB + qbase + r + half16 * 8;
      out[row * Hh + nc * 16 + l16] = accO[nc][r] / lrow[r];
    }
}

// ---------------------------------------------------------------------------
extern "C" void kernel_launch(void* const* d_in, const int* in_sizes, int n_in,
                              void* d_out, int out_size, void* d_ws, size_t ws_size,
                              hipStream_t stream) {
  const float* x  = (const float*)d_in[0];
  const float* Wk = (const float*)d_in[1];
  const float* Wq = (const float*)d_in[2];
  const float* Wv = (const float*)d_in[3];

  const size_t NH = (size_t)Bb * Tt * Hh;          // 1M elems
  _Float16* qb  = (_Float16*)d_ws;                 // 2 MB
  _Float16* kb  = qb + NH;                         // 2 MB
  _Float16* vTb = kb + NH;                         // 2 MB, [B][H][T]
  float* out = (float*)d_out;

  // 1024 row-tiles, 8 waves/block -> 128 blocks
  proj_qkv<<<(Bb * Tt) / (16 * 8), 256, 0, stream>>>(x, Wk, Wq, Wv, qb, kb, vTb);
  // 1024 query tiles, 8 waves/block -> 128 blocks, barrier-free
  attn_flash<<<(Bb * (Tt / 16)) / 8, 256, 0, stream>>>(qb, kb, vTb, out);
}
